// FutureDecoder_36696200577060
// MI455X (gfx1250) — matrix operand order repaired and checked by reading, not compiled, not run on hardware
//
#include <hip/hip_runtime.h>
#include <stdint.h>

#define DECN     128
#define KTOT     288     // 160 padded-x cols + 128 h cols
#define KXPAD    160
#define XPAD     296     // LDS row stride (bf16 elems), 16B-aligned rows
#define HPAD     136
#define ROWS     64      // rows per block
#define THREADS  512     // 16 waves
#define SMSTRIDE 1536    // per-vehicle small-param stride (f32)

typedef __bf16 v16bf __attribute__((ext_vector_type(16)));
typedef __bf16 v8bf  __attribute__((ext_vector_type(8)));
typedef float  v8f   __attribute__((ext_vector_type(8)));

union V16U { v16bf v; v8bf h[2]; };

struct AllPtrs { const float* p[4][17]; };

__device__ __forceinline__ uint32_t hashu(uint32_t x){
  x ^= x >> 16; x *= 0x7feb352du; x ^= x >> 15; x *= 0x846ca68bu; x ^= x >> 16; return x;
}
__device__ __forceinline__ float u01(uint32_t h){
  return (float)(h >> 8) * (1.0f/16777216.0f) + 1e-7f;
}
__device__ __forceinline__ float sigmoidf(float x){ return 1.0f/(1.0f + __expf(-x)); }

// ---------------------------------------------------------------------------
// Prep: weights -> bf16, K-major (transposed), padded; pack small params.
// Leaf order (sorted keys): Wa0 Wg1 Wh2 Wml3 Wmt4 Wr5 Wsl6 Wst7 Wx8 b9 ba10
//                           bg11 bml12 bmt13 br14 bsl15 bst16
// ---------------------------------------------------------------------------
__global__ __launch_bounds__(256) void prep_weights_kernel(
    AllPtrs ap, __bf16* __restrict__ wxhT, __bf16* __restrict__ wgT,
    float* __restrict__ smalls)
{
  const int v = blockIdx.z;
  const int nAtab[4] = {8, 6, 2, 2};
  const int nA = nAtab[v];
  const float* Wx = ap.p[v][8];
  const float* Wh = ap.p[v][2];
  const float* Wg = ap.p[v][1];
  const int tid = blockIdx.x * blockDim.x + threadIdx.x;
  const int stride = gridDim.x * blockDim.x;

  // W^T[512][288]: cols 0..7 = action rows (zero-padded), 8..135 = enc_h rows,
  // 136..159 = zero pad, 160..287 = Wh rows.
  for (int idx = tid; idx < 512 * KTOT; idx += stride){
    int n = idx / KTOT, kk = idx - n * KTOT;
    float val = 0.f;
    if (kk < 8)          { if (kk < nA) val = Wx[kk * 512 + n]; }
    else if (kk < 136)   val = Wx[(nA + kk - 8) * 512 + n];
    else if (kk < KXPAD) val = 0.f;
    else                 val = Wh[(kk - KXPAD) * 512 + n];
    wxhT[((size_t)v * 512 + n) * KTOT + kk] = (__bf16)val;
  }
  // Wg^T[32][128] (30 valid rows)
  for (int idx = tid; idx < 32 * DECN; idx += stride){
    int n = idx / DECN, k = idx - n * DECN;
    float val = (n < 30) ? Wg[k * 30 + n] : 0.f;
    wgT[((size_t)v * 32 + n) * DECN + k] = (__bf16)val;
  }
  // smalls: [0,512)=b  [512,544)=bg(pad32)  [544,1444)=6x(30x5)  [1444,1474)=6x5 bias
  float* sm = smalls + (size_t)v * SMSTRIDE;
  const int leafW[6] = {0, 5, 3, 6, 4, 7};     // Wa Wr Wml Wsl Wmt Wst
  const int leafB[6] = {10, 14, 12, 15, 13, 16};
  for (int idx = tid; idx < SMSTRIDE; idx += stride){
    float val = 0.f;
    if (idx < 512)       val = ap.p[v][9][idx];
    else if (idx < 544)  { int j = idx - 512; val = (j < 30) ? ap.p[v][11][j] : 0.f; }
    else if (idx < 1444) { int q = idx - 544; int hx = q / 150; val = ap.p[v][leafW[hx]][q - hx * 150]; }
    else if (idx < 1474) { int q = idx - 1444; int hx = q / 5; val = ap.p[v][leafB[hx]][q - hx * 5]; }
    sm[idx] = val;
  }
}

__global__ __launch_bounds__(256) void prep_state_kernel(
    const float* __restrict__ sh, const float* __restrict__ sc,
    const float* __restrict__ cm, const float* __restrict__ cy,
    const float* __restrict__ cf, const float* __restrict__ cj,
    __bf16* __restrict__ encbf, float* __restrict__ hbuf, float* __restrict__ cbuf,
    float* __restrict__ act0, int B, int T)
{
  const int tid = blockIdx.x * blockDim.x + threadIdx.x;
  const int stride = gridDim.x * blockDim.x;
  const int n = B * DECN;
  for (int i = tid; i < n; i += stride){
    float hv = sh[i], cv = sc[i];
    encbf[i] = (__bf16)hv;
    #pragma unroll
    for (int v = 0; v < 4; v++){
      hbuf[(size_t)v * n + i] = hv;
      cbuf[(size_t)v * n + i] = cv;
    }
  }
  for (int b = tid; b < B; b += stride){
    size_t src = (size_t)b * T * 2;
    act0[((size_t)0 * B + b) * 2 + 0] = cm[src]; act0[((size_t)0 * B + b) * 2 + 1] = cm[src + 1];
    act0[((size_t)1 * B + b) * 2 + 0] = cy[src]; act0[((size_t)1 * B + b) * 2 + 1] = cy[src + 1];
    act0[((size_t)2 * B + b) * 2 + 0] = cf[src]; act0[((size_t)2 * B + b) * 2 + 1] = cf[src + 1];
    act0[((size_t)3 * B + b) * 2 + 0] = cj[src]; act0[((size_t)3 * B + b) * 2 + 1] = cj[src + 1];
  }
}

// ---------------------------------------------------------------------------
// One timestep, all four vehicles. grid = (B/64, 4), 512 threads = 16 waves.
// ---------------------------------------------------------------------------
__global__ __launch_bounds__(THREADS) void step_kernel(
    const __bf16* __restrict__ wxhT, const __bf16* __restrict__ wgT,
    const float*  __restrict__ smalls, const __bf16* __restrict__ encbf,
    float* __restrict__ hbuf, float* __restrict__ cbuf,
    const float* __restrict__ aPrev, float* __restrict__ aNext,
    const float* __restrict__ cm, const float* __restrict__ cy,
    const float* __restrict__ cf, const float* __restrict__ cj,
    float* __restrict__ out, int B, int T, int t)
{
  __shared__ __bf16 Xs[ROWS][XPAD];   // [x(160) | h(128)] bf16 tile   (37.9 KB)
  __shared__ __bf16 Hs[ROWS][HPAD];   // new h, bf16                   (17.4 KB)
  __shared__ float  Os[ROWS][36];     // GEMM2 output o-vector         ( 9.2 KB)
  __shared__ float  Hd[ROWS][32];     // raw head pre-activations      ( 8.2 KB)

  const int v    = blockIdx.y;
  const int row0 = blockIdx.x * ROWS;
  const int tid  = threadIdx.x;

  // ---- build X tile -------------------------------------------------------
  for (int idx = tid; idx < ROWS * DECN; idx += THREADS){
    int r = idx >> 7, k = idx & 127;
    Xs[r][8 + k]     = encbf[(size_t)(row0 + r) * DECN + k];
    Xs[r][KXPAD + k] = (__bf16)hbuf[((size_t)v * B + row0 + r) * DECN + k];
  }
  for (int idx = tid; idx < ROWS * 24; idx += THREADS){
    int r = idx / 24, k = idx - r * 24;
    Xs[r][136 + k] = (__bf16)0.f;
  }
  if (tid < ROWS){
    int bg = row0 + tid;
    size_t co = ((size_t)bg * T + t) * 2;
    float am0 = aPrev[((size_t)0 * B + bg) * 2 + 0], am1 = aPrev[((size_t)0 * B + bg) * 2 + 1];
    float ay0 = aPrev[((size_t)1 * B + bg) * 2 + 0], ay1 = aPrev[((size_t)1 * B + bg) * 2 + 1];
    float af0 = aPrev[((size_t)2 * B + bg) * 2 + 0], af1 = aPrev[((size_t)2 * B + bg) * 2 + 1];
    float aj0 = aPrev[((size_t)3 * B + bg) * 2 + 0], aj1 = aPrev[((size_t)3 * B + bg) * 2 + 1];
    float tm0 = cm[co], tm1 = cm[co + 1], ty0 = cy[co], ty1 = cy[co + 1];
    float tf0 = cf[co], tf1 = cf[co + 1], tj0 = cj[co], tj1 = cj[co + 1];
    am0 = (fabsf(am0 - tm0) < 0.3f) ? am0 : tm0;  am1 = (fabsf(am1 - tm1) < 0.3f) ? am1 : tm1;
    ay0 = (fabsf(ay0 - ty0) < 0.3f) ? ay0 : ty0;  ay1 = (fabsf(ay1 - ty1) < 0.3f) ? ay1 : ty1;
    af0 = (fabsf(af0 - tf0) < 0.3f) ? af0 : tf0;  af1 = (fabsf(af1 - tf1) < 0.3f) ? af1 : tf1;
    aj0 = (fabsf(aj0 - tj0) < 0.3f) ? aj0 : tj0;  aj1 = (fabsf(aj1 - tj1) < 0.3f) ? aj1 : tj1;
    float a8[8] = {0.f, 0.f, 0.f, 0.f, 0.f, 0.f, 0.f, 0.f};
    if (v == 0){ a8[0]=am0; a8[1]=am1; a8[2]=ay0; a8[3]=ay1; a8[4]=af0; a8[5]=af1; a8[6]=aj0; a8[7]=aj1; }
    else if (v == 1){ a8[0]=am0; a8[1]=am1; a8[2]=ay0; a8[3]=ay1; a8[4]=aj0; a8[5]=aj1; }
    else if (v == 2){ a8[0]=af0; a8[1]=af1; }
    else            { a8[0]=aj0; a8[1]=aj1; }
    #pragma unroll
    for (int k = 0; k < 8; k++) Xs[tid][k] = (__bf16)a8[k];
  }
  __syncthreads();

  // ---- GEMM1: Z[64x512] = [X|H][64x288] @ W^T, gate-aligned wave tiling ---
  const int lane = tid & 31;
  const int w    = tid >> 5;     // 0..15
  const int wr   = w >> 2;       // row-wave 0..3 (16 rows each)
  const int wc   = w & 3;        // hidden-col wave 0..3 (32 hid cols each)
  const int ln   = lane & 15;
  const int hi   = lane >> 4;

  v8f acc[8];                    // [gate 0..3][coltile 0..1]
  #pragma unroll
  for (int g = 0; g < 4; g++)
    #pragma unroll
    for (int cc = 0; cc < 2; cc++){
      int n = 128 * g + 32 * wc + 16 * cc + ln;
      float bv = smalls[(size_t)v * SMSTRIDE + n];
      v8f tmp;
      #pragma unroll
      for (int e = 0; e < 8; e++) tmp[e] = bv;
      acc[g * 2 + cc] = tmp;
    }

  const __bf16* WT = wxhT + (size_t)v * 512 * KTOT;
  const int arow = 16 * wr + ln;
  #pragma unroll 1
  for (int i = 0; i < 9; i++){
    int kbA = 32 * i + (hi ? 16 : 0);
    V16U a;
    a.h[0] = *(const v8bf*)&Xs[arow][kbA];
    a.h[1] = *(const v8bf*)&Xs[arow][kbA + 8];
    int kbB = 32 * i + (hi ? 16 : 0);
    if (i + 1 < 9)  // warm L2/L0 for next k-panel of this wave's first column
      __builtin_prefetch(WT + (size_t)(32 * wc + ln) * KTOT + 32 * (i + 1), 0, 1);
    #pragma unroll
    for (int g = 0; g < 4; g++){
      #pragma unroll
      for (int cc = 0; cc < 2; cc++){
        int n = 128 * g + 32 * wc + 16 * cc + ln;
        const v16bf* pb = (const v16bf*)(WT + (size_t)n * KTOT + kbB);
        acc[g * 2 + cc] = __builtin_amdgcn_wmma_f32_16x16x32_bf16(
            false, a.v, false, *pb, (short)0, acc[g * 2 + cc], false, false);
      }
    }
  }

  // ---- LSTM elementwise (all gates resident in this wave's registers) -----
  #pragma unroll
  for (int cc = 0; cc < 2; cc++){
    int hidc = 32 * wc + 16 * cc + ln;
    #pragma unroll
    for (int rr = 0; rr < 8; rr++){
      int rloc = 16 * wr + rr + 8 * hi;
      size_t gidx = ((size_t)v * B + row0 + rloc) * DECN + hidc;
      float cold = cbuf[gidx];
      float iv = sigmoidf(acc[0 + cc][rr]);
      float fv = sigmoidf(acc[2 + cc][rr]);
      float gv = tanhf(acc[4 + cc][rr]);
      float ov = sigmoidf(acc[6 + cc][rr]);
      float cn = fv * cold + iv * gv;
      float hn = ov * tanhf(cn);
      cbuf[gidx] = cn;
      hbuf[gidx] = hn;
      Hs[rloc][hidc] = (__bf16)hn;
    }
  }
  __syncthreads();

  // ---- GEMM2: O[64x32] = Hnew[64x128] @ Wg^T (30 valid cols) --------------
  if (w < 4){
    v8f oacc[2];
    #pragma unroll
    for (int cc = 0; cc < 2; cc++){
      int n = 16 * cc + ln;
      float bgv = (n < 30) ? smalls[(size_t)v * SMSTRIDE + 512 + n] : 0.f;
      v8f tmp;
      #pragma unroll
      for (int e = 0; e < 8; e++) tmp[e] = bgv;
      oacc[cc] = tmp;
    }
    const __bf16* GT = wgT + (size_t)v * 32 * DECN;
    const int arow2 = 16 * w + ln;
    #pragma unroll
    for (int i = 0; i < 4; i++){
      int kbA = 32 * i + (hi ? 16 : 0);
      V16U a;
      a.h[0] = *(const v8bf*)&Hs[arow2][kbA];
      a.h[1] = *(const v8bf*)&Hs[arow2][kbA + 8];
      int kbB = 32 * i + (hi ? 16 : 0);
      #pragma unroll
      for (int cc = 0; cc < 2; cc++){
        const v16bf* pb = (const v16bf*)(GT + (size_t)(16 * cc + ln) * DECN + kbB);
        oacc[cc] = __builtin_amdgcn_wmma_f32_16x16x32_bf16(
            false, a.v, false, *pb, (short)0, oacc[cc], false, false);
      }
    }
    #pragma unroll
    for (int cc = 0; cc < 2; cc++)
      #pragma unroll
      for (int rr = 0; rr < 8; rr++)
        Os[16 * w + rr + 8 * hi][16 * cc + ln] = oacc[cc][rr];
  }
  __syncthreads();

  // ---- heads part A: hd[row][hx][j] = o @ W_hx + b_hx (384 threads) -------
  if (tid < ROWS * 6){
    const int rw = tid / 6, hx = tid - 6 * (tid / 6);
    const float* sm = smalls + (size_t)v * SMSTRIDE;
    const float* Wp = sm + 544 + hx * 150;
    float s0 = sm[1444 + hx * 5 + 0], s1 = sm[1444 + hx * 5 + 1];
    float s2 = sm[1444 + hx * 5 + 2], s3 = sm[1444 + hx * 5 + 3];
    float s4 = sm[1444 + hx * 5 + 4];
    #pragma unroll 5
    for (int k = 0; k < 30; k++){
      float ov = Os[rw][k];
      s0 += ov * Wp[k * 5 + 0];
      s1 += ov * Wp[k * 5 + 1];
      s2 += ov * Wp[k * 5 + 2];
      s3 += ov * Wp[k * 5 + 3];
      s4 += ov * Wp[k * 5 + 4];
    }
    Hd[rw][hx * 5 + 0] = s0; Hd[rw][hx * 5 + 1] = s1; Hd[rw][hx * 5 + 2] = s2;
    Hd[rw][hx * 5 + 3] = s3; Hd[rw][hx * 5 + 4] = s4;
  }
  __syncthreads();

  // ---- heads part B: activations + gp write + sampling (1 thread/row) -----
  if (tid < ROWS){
    const int r = tid, bg = row0 + r;
    float al[5], rh[5], sl[5], st[5], ml[5], mt[5];
    float mx = Hd[r][0];
    #pragma unroll
    for (int j = 1; j < 5; j++) mx = fmaxf(mx, Hd[r][j]);
    float se = 0.f;
    #pragma unroll
    for (int j = 0; j < 5; j++){ al[j] = __expf(Hd[r][j] - mx); se += al[j]; }
    float inv = 1.f / se;
    #pragma unroll
    for (int j = 0; j < 5; j++){
      al[j] *= inv;
      rh[j] = tanhf(Hd[r][5 + j]);
      ml[j] = Hd[r][10 + j];
      sl[j] = __expf(Hd[r][15 + j]);
      mt[j] = Hd[r][20 + j];
      st[j] = __expf(Hd[r][25 + j]);
    }
    float* og = out + ((size_t)v * B * T + (size_t)bg * T + t) * 30;
    #pragma unroll
    for (int j = 0; j < 5; j++){
      og[j] = al[j]; og[5 + j] = rh[j]; og[10 + j] = ml[j];
      og[15 + j] = sl[j]; og[20 + j] = mt[j]; og[25 + j] = st[j];
    }
    // deterministic gumbel-argmax + Box-Muller normals
    uint32_t s0 = hashu((uint32_t)(t * 4 + v) * 0x9E3779B9u ^ (uint32_t)bg * 0x85EBCA6Bu);
    int comp = 0; float best = -1e30f;
    #pragma unroll
    for (int j = 0; j < 5; j++){
      float u = u01(hashu(s0 + 0x68BC21EBu * (uint32_t)(j + 1)));
      float gum = -__logf(-__logf(u));
      float sc2 = __logf(al[j] + 1e-20f) + gum;
      if (sc2 > best){ best = sc2; comp = j; }
    }
    float rho = 0.f, mlc = 0.f, slc = 0.f, mtc = 0.f, stc = 0.f;
    #pragma unroll
    for (int j = 0; j < 5; j++)
      if (j == comp){ rho = rh[j]; mlc = ml[j]; slc = sl[j]; mtc = mt[j]; stc = st[j]; }
    float u1 = u01(hashu(s0 ^ 0xD1B54A33u));
    float u2 = u01(hashu(s0 + 0x94D049BBu));
    float rad = __fsqrt_rn(fmaxf(0.f, -2.f * __logf(u1)));
    float z1  = rad * __cosf(6.28318530718f * u2);
    float zz2 = rad * __sinf(6.28318530718f * u2);
    float alon = mlc + slc * z1;
    float alat = mtc + stc * (rho * z1 + __fsqrt_rn(fmaxf(0.f, 1.f - rho * rho)) * zz2);
    alon = fminf(5.f, fmaxf(-5.f, alon));
    alat = fminf(5.f, fmaxf(-5.f, alat));
    aNext[((size_t)v * B + bg) * 2 + 0] = alon;
    aNext[((size_t)v * B + bg) * 2 + 1] = alat;
  }
}

// ---------------------------------------------------------------------------
extern "C" void kernel_launch(void* const* d_in, const int* in_sizes, int n_in,
                              void* d_out, int out_size, void* d_ws, size_t ws_size,
                              hipStream_t stream)
{
  const int B = in_sizes[4] / DECN;
  const int T = in_sizes[0] / (B * 2);

  const float* cm = (const float*)d_in[0];
  const float* cy = (const float*)d_in[1];
  const float* cf = (const float*)d_in[2];
  const float* cj = (const float*)d_in[3];
  const float* sh = (const float*)d_in[4];
  const float* sc = (const float*)d_in[5];

  // params flatten: sorted top-level keys f,fadj,m,y; 17 sorted leaves each
  AllPtrs ap;
  const int vmap[4] = {2, 3, 0, 1};   // our order m,y,f,fadj -> flatten slot
  for (int v = 0; v < 4; v++)
    for (int l = 0; l < 17; l++)
      ap.p[v][l] = (const float*)d_in[6 + 17 * vmap[v] + l];

  char* ws = (char*)d_ws;
  size_t off = 0;
  auto take = [&](size_t bytes) -> void* {
    void* p = ws + off;
    off = (off + bytes + 255) & ~(size_t)255;
    return p;
  };
  __bf16* wxhT  = (__bf16*)take((size_t)4 * 512 * KTOT * 2);
  __bf16* wgT   = (__bf16*)take((size_t)4 * 32 * DECN * 2);
  float*  smalls= (float*)take((size_t)4 * SMSTRIDE * 4);
  __bf16* encbf = (__bf16*)take((size_t)B * DECN * 2);
  float*  hbuf  = (float*)take((size_t)4 * B * DECN * 4);
  float*  cbuf  = (float*)take((size_t)4 * B * DECN * 4);
  float*  act   = (float*)take((size_t)2 * 4 * B * 2 * 4);
  (void)ws_size; (void)n_in; (void)out_size;

  prep_weights_kernel<<<dim3(64, 1, 4), 256, 0, stream>>>(ap, wxhT, wgT, smalls);
  prep_state_kernel<<<dim3(256), 256, 0, stream>>>(sh, sc, cm, cy, cf, cj,
                                                   encbf, hbuf, cbuf, act, B, T);

  for (int t = 0; t < T; t++){
    const float* aP = act + (size_t)(t & 1) * 4 * B * 2;
    float*       aN = act + (size_t)((t + 1) & 1) * 4 * B * 2;
    step_kernel<<<dim3(B / ROWS, 4), THREADS, 0, stream>>>(
        wxhT, wgT, smalls, encbf, hbuf, cbuf, aP, aN,
        cm, cy, cf, cj, (float*)d_out, B, T, t);
  }
}
